// SwinTransformerBlock_75204877353457
// MI455X (gfx1250) — compile-verified, matrix operand order checked by reading, not run on hardware
//
#include <hip/hip_runtime.h>
#include <hip/hip_bf16.h>
#include <math.h>

// ---------------------------------------------------------------------------
// Swin Transformer block for MI455X (gfx1250), wave32 + WMMA f32_16x16x32_f16.
// All WMMA operands are staged in LDS K-contiguous so every fragment load is
// two ds_load_b128 per lane. Contiguous activation staging uses the CDNA5
// async global->LDS path (ASYNCcnt) when the toolchain exposes it.
// ---------------------------------------------------------------------------

typedef __attribute__((ext_vector_type(16))) _Float16 v16h;
typedef __attribute__((ext_vector_type(8)))  _Float16 v8h;
typedef __attribute__((ext_vector_type(8)))  float    v8f;
typedef __attribute__((ext_vector_type(4)))  int      v4i;

#define HH 56
#define WW_ 56
#define CC 192
#define WS_ 7
#define SS_ 3
#define NH_ 6
#define HD_ 32
#define NTOK 200704      // 64 * 56 * 56
#define NWIN 4096        // 64 batches * 64 windows
#define HIDDEN_ 768

#if defined(__gfx1250__) &&                                         \
    __has_builtin(__builtin_amdgcn_global_load_async_to_lds_b128) && \
    __has_builtin(__builtin_amdgcn_s_wait_asynccnt)
#define HAVE_ASYNC_LDS 1
typedef __attribute__((address_space(1))) v4i* global_v4i_ptr;
typedef __attribute__((address_space(3))) v4i* lds_v4i_ptr;
#endif

// 16-byte global -> LDS copy; async (tracked by ASYNCcnt) when available.
__device__ __forceinline__ void copy16_g2l(const _Float16* g, _Float16* l) {
#ifdef HAVE_ASYNC_LDS
  __builtin_amdgcn_global_load_async_to_lds_b128(
      (global_v4i_ptr)(v4i*)g, (lds_v4i_ptr)(v4i*)l, 0, 0);
#else
  *(v8h*)l = *(const v8h*)g;
#endif
}
__device__ __forceinline__ void copy_join() {
#ifdef HAVE_ASYNC_LDS
  __builtin_amdgcn_s_wait_asynccnt(0);
#endif
}

__device__ __forceinline__ v8f zero8() {
  v8f z = {0.f, 0.f, 0.f, 0.f, 0.f, 0.f, 0.f, 0.f};
  return z;
}

__device__ __forceinline__ v8f wmma16(v16h a, v16h b, v8f c) {
  // D = A(16x32) * B(32x16) + C, f32 accumulate  ->  v_wmma_f32_16x16x32_f16
  return __builtin_amdgcn_wmma_f32_16x16x32_f16(false, a, false, b, (short)0, c,
                                                false, false);
}

// Fragment load for 16-bit WMMA operands from K-contiguous storage.
// For A operands: p[row][k] row-major, row = M.  For B operands the source
// must be stored N-major (Bt[n][k]), i.e. K-contiguous per output column.
// Per the ISA (7.12.2) lane (r = lane&15, half = lane>>4) needs
// k = k0 + [half*8, half*8+8) and k0 + [16+half*8, ...): two b128 loads.
__device__ __forceinline__ v16h load_frag(const _Float16* p, int stride,
                                          int row0, int k0, int lane) {
  const int r = lane & 15, half = lane >> 4;
  const _Float16* base = p + (row0 + r) * stride + k0 + half * 8;
  const v8h lo = *(const v8h*)base;
  const v8h hi = *(const v8h*)(base + 16);
  return __builtin_shufflevector(lo, hi, 0, 1, 2, 3, 4, 5, 6, 7, 8, 9, 10, 11,
                                 12, 13, 14, 15);
}

// ---------------------------------------------------------------------------
// Kernel 0: f32 -> f16 weight conversion
// ---------------------------------------------------------------------------
__global__ void cvt_f16_kernel(const float* __restrict__ src,
                               _Float16* __restrict__ dst, int n) {
  int i = blockIdx.x * 256 + threadIdx.x;
  if (i < n) dst[i] = (_Float16)src[i];
}

// ---------------------------------------------------------------------------
// Kernel 1/4: LayerNorm over C=192, one wave per token, output f16
// ---------------------------------------------------------------------------
__global__ __launch_bounds__(256)
void ln_f16_kernel(const float* __restrict__ x, const float* __restrict__ w,
                   const float* __restrict__ b, _Float16* __restrict__ out) {
  const int lane = threadIdx.x & 31;
  const int tok = blockIdx.x * 8 + (threadIdx.x >> 5);
  if (tok >= NTOK) return;
  const float* row = x + (size_t)tok * CC;
  float v[6], s = 0.f, ss = 0.f;
#pragma unroll
  for (int i = 0; i < 6; ++i) {
    v[i] = row[i * 32 + lane];
    s += v[i];
    ss += v[i] * v[i];
  }
#pragma unroll
  for (int off = 16; off; off >>= 1) {
    s += __shfl_xor(s, off, 32);
    ss += __shfl_xor(ss, off, 32);
  }
  const float mu = s * (1.f / CC);
  const float var = ss * (1.f / CC) - mu * mu;
  const float rs = rsqrtf(var + 1e-5f);
  _Float16* orow = out + (size_t)tok * CC;
#pragma unroll
  for (int i = 0; i < 6; ++i) {
    const int c = i * 32 + lane;
    orow[c] = (_Float16)((v[i] - mu) * rs * w[c] + b[c]);
  }
}

// ---------------------------------------------------------------------------
// Kernel 2: fused window attention (QKV GEMM + bias/mask + softmax + P@V)
//   one workgroup (8 waves) per window; LDS = 60 KB
// ---------------------------------------------------------------------------
__global__ __launch_bounds__(256)
void attn_kernel(const _Float16* __restrict__ xn, const _Float16* __restrict__ wq,
                 const float* __restrict__ qkv_b, const float* __restrict__ rpb,
                 _Float16* __restrict__ o_ws) {
  __shared__ __attribute__((aligned(16))) _Float16 As[64 * 192];  // window acts
  __shared__ __attribute__((aligned(16))) _Float16 Bs[64 * 64];   // Wt slice / P
  __shared__ __attribute__((aligned(16))) _Float16 Qh[64 * 32];   // q (scaled)
  __shared__ __attribute__((aligned(16))) _Float16 Kh[64 * 32];   // k
  __shared__ __attribute__((aligned(16))) _Float16 Vt[32 * 64];   // v transposed
  __shared__ __attribute__((aligned(16))) float    Sb[64 * 64];   // logits

  const int tid = threadIdx.x, lane = tid & 31, wave = tid >> 5;
  const int win = blockIdx.x;
  const int b = win >> 6, wid = win & 63, wy = wid >> 3, wx = wid & 7;

  // gather LN'd activations with cyclic shift; rows >= 49 zero-padded.
  // async global->LDS, 8 halves (16B) per copy (row length 192 = 24 chunks).
  for (int ch = tid; ch < 64 * 24; ch += 256) {
    const int r = ch / 24, c8 = (ch % 24) * 8;
    if (r < 49) {
      const int i = r / 7, j = r % 7;
      const int hh = (wy * 7 + i + SS_) % HH;
      const int ww = (wx * 7 + j + SS_) % WW_;
      copy16_g2l(&xn[((size_t)b * 3136 + hh * 56 + ww) * CC + c8],
                 &As[r * 192 + c8]);
    } else {
      v8h z = {};
      *(v8h*)&As[r * 192 + c8] = z;
    }
  }
  copy_join();
  __syncthreads();

  const float scale = 0.17677669529663687f;  // 1/sqrt(32)

  for (int h = 0; h < NH_; ++h) {
    // ---- QKV GEMM: (64x192) x (192x96) -> Qh|Kh|Vt ----
    v8f acc[3] = {zero8(), zero8(), zero8()};
    for (int kk = 0; kk < 192; kk += 32) {
      __syncthreads();
      // stage W k-slice TRANSPOSED: Bs[c*32 + kr] (coalesced global reads)
      for (int idx = tid; idx < 32 * 96; idx += 256) {
        const int kr = idx / 96, c = idx % 96;
        const int sct = c >> 5, d = c & 31;
        Bs[c * 32 + kr] = wq[(size_t)(kk + kr) * 576 + sct * 192 + h * 32 + d];
      }
      if (kk + 32 < 192)
        __builtin_prefetch(&wq[(size_t)(kk + 32) * 576 + h * 32], 0, 1);
      __syncthreads();
#pragma unroll
      for (int j = 0; j < 3; ++j) {
        const int t = wave + 8 * j;          // 24 tiles over 64x96
        const int tm = t / 6, tn = t % 6;
        v16h a = load_frag(As, 192, tm * 16, kk, lane);
        v16h bb = load_frag(Bs, 32, tn * 16, 0, lane);
        acc[j] = wmma16(a, bb, acc[j]);
      }
    }
#pragma unroll
    for (int j = 0; j < 3; ++j) {
      const int t = wave + 8 * j;
      const int tm = t / 6, tn = t % 6;
#pragma unroll
      for (int e = 0; e < 8; ++e) {
        const int m = tm * 16 + e + 8 * (lane >> 4);
        const int n = tn * 16 + (lane & 15);
        const int sct = n >> 5, d = n & 31;
        const float val = acc[j][e] + qkv_b[sct * 192 + h * 32 + d];
        if (sct == 0)      Qh[m * 32 + d] = (_Float16)(val * scale);
        else if (sct == 1) Kh[m * 32 + d] = (_Float16)val;
        else               Vt[d * 64 + m] = (_Float16)val;
      }
    }
    __syncthreads();

    // ---- S = q @ kT  (64x64, K=32) + rel-pos bias + shift mask ----
    v8f sacc[2] = {zero8(), zero8()};
#pragma unroll
    for (int j = 0; j < 2; ++j) {
      const int t = wave * 2 + j;            // 16 tiles over 64x64
      const int tm = t >> 2, tn = t & 3;
      v16h a = load_frag(Qh, 32, tm * 16, 0, lane);
      v16h bb = load_frag(Kh, 32, tn * 16, 0, lane);  // Kh rows are kT columns
      sacc[j] = wmma16(a, bb, sacc[j]);
    }
#pragma unroll
    for (int j = 0; j < 2; ++j) {
      const int t = wave * 2 + j;
      const int tm = t >> 2, tn = t & 3;
#pragma unroll
      for (int e = 0; e < 8; ++e) {
        const int m = tm * 16 + e + 8 * (lane >> 4);
        const int n = tn * 16 + (lane & 15);
        float sval = -1e30f;
        if (m < 49 && n < 49) {
          const int i1 = m / 7, j1 = m % 7, i2 = n / 7, j2 = n % 7;
          const int ridx = (i1 - i2 + 6) * 13 + (j1 - j2 + 6);
          const float bias = rpb[ridx * NH_ + h];
          const int hu1 = wy * 7 + i1, wu1 = wx * 7 + j1;
          const int hu2 = wy * 7 + i2, wu2 = wx * 7 + j2;
          const int rh1 = hu1 < 49 ? 0 : (hu1 < 53 ? 1 : 2);
          const int rh2 = hu2 < 49 ? 0 : (hu2 < 53 ? 1 : 2);
          const int rw1 = wu1 < 49 ? 0 : (wu1 < 53 ? 1 : 2);
          const int rw2 = wu2 < 49 ? 0 : (wu2 < 53 ? 1 : 2);
          const float maskv = (rh1 == rh2 && rw1 == rw2) ? 0.f : -100.f;
          sval = sacc[j][e] + bias + maskv;
        }
        Sb[m * 64 + n] = sval;
      }
    }
    __syncthreads();

    // ---- row softmax -> P (f16, 64x64, padded zeros) stored in Bs ----
    if (tid < 64) {
      const int r = tid;
      if (r < 49) {
        float mx = -1e30f;
        for (int c = 0; c < 49; ++c) mx = fmaxf(mx, Sb[r * 64 + c]);
        float sum = 0.f;
        for (int c = 0; c < 49; ++c) sum += __expf(Sb[r * 64 + c] - mx);
        const float inv = 1.f / sum;
        for (int c = 0; c < 64; ++c) {
          const float p = (c < 49) ? __expf(Sb[r * 64 + c] - mx) * inv : 0.f;
          Bs[r * 64 + c] = (_Float16)p;
        }
      } else {
        for (int c = 0; c < 64; ++c) Bs[r * 64 + c] = (_Float16)0.f;
      }
    }
    __syncthreads();

    // ---- O = P(64x64) @ v(64x32) ----
    v8f oacc = zero8();
    const int tm = wave >> 1, tn = wave & 1;   // 8 tiles over 64x32
#pragma unroll
    for (int kk = 0; kk < 64; kk += 32) {
      v16h a = load_frag(Bs, 64, tm * 16, kk, lane);
      v16h bb = load_frag(Vt, 64, tn * 16, kk, lane);  // Vt rows = out dims
      oacc = wmma16(a, bb, oacc);
    }
#pragma unroll
    for (int e = 0; e < 8; ++e) {
      const int m = tm * 16 + e + 8 * (lane >> 4);
      const int n = tn * 16 + (lane & 15);
      o_ws[((size_t)(win * NH_ + h) * 64 + m) * HD_ + n] = (_Float16)oacc[e];
    }
    __syncthreads();
  }
}

// ---------------------------------------------------------------------------
// Kernel 3: output projection + bias + reverse-shift scatter + residual
// ---------------------------------------------------------------------------
__global__ __launch_bounds__(256)
void proj_kernel(const _Float16* __restrict__ o_ws, const _Float16* __restrict__ wp,
                 const float* __restrict__ proj_b, const float* __restrict__ x,
                 float* __restrict__ y1) {
  __shared__ __attribute__((aligned(16))) _Float16 As[64 * 192];
  __shared__ __attribute__((aligned(16))) _Float16 Bs[64 * 32];  // transposed
  const int tid = threadIdx.x, lane = tid & 31, wave = tid >> 5;
  const int win = blockIdx.x;
  const int b = win >> 6, wid = win & 63, wy = wid >> 3, wx = wid & 7;

  // gather attention output (concat heads); async 16B copies
  for (int ch = tid; ch < 64 * 24; ch += 256) {
    const int r = ch / 24, c8 = (ch % 24) * 8;
    const int h = c8 >> 5, d = c8 & 31;
    if (r < 49) {
      copy16_g2l(&o_ws[((size_t)(win * NH_ + h) * 64 + r) * HD_ + d],
                 &As[r * 192 + c8]);
    } else {
      v8h z = {};
      *(v8h*)&As[r * 192 + c8] = z;
    }
  }
  copy_join();
  __syncthreads();

  for (int ct = 0; ct < 3; ++ct) {   // 3 column tiles of 64
    v8f acc[2] = {zero8(), zero8()};
    for (int kk = 0; kk < 192; kk += 32) {
      __syncthreads();
      for (int idx = tid; idx < 32 * 64; idx += 256) {
        const int kr = idx >> 6, c = idx & 63;
        Bs[c * 32 + kr] = wp[(size_t)(kk + kr) * CC + ct * 64 + c];
      }
      __syncthreads();
#pragma unroll
      for (int j = 0; j < 2; ++j) {
        const int t = wave + 8 * j;         // 16 tiles over 64x64
        const int tm = t >> 2, tn = t & 3;
        v16h a = load_frag(As, 192, tm * 16, kk, lane);
        v16h bb = load_frag(Bs, 32, tn * 16, 0, lane);
        acc[j] = wmma16(a, bb, acc[j]);
      }
    }
#pragma unroll
    for (int j = 0; j < 2; ++j) {
      const int t = wave + 8 * j;
      const int tm = t >> 2, tn = t & 3;
#pragma unroll
      for (int e = 0; e < 8; ++e) {
        const int m = tm * 16 + e + 8 * (lane >> 4);
        const int n = ct * 64 + tn * 16 + (lane & 15);
        if (m < 49) {
          const int i = m / 7, j2 = m % 7;
          const int hh = (wy * 7 + i + SS_) % HH;
          const int ww = (wx * 7 + j2 + SS_) % WW_;
          const size_t o = ((size_t)b * 3136 + hh * 56 + ww) * CC + n;
          y1[o] = x[o] + acc[j][e] + proj_b[n];
        }
      }
    }
  }
}

// ---------------------------------------------------------------------------
// Kernel 5: fused MLP: fc1 + exact GELU + fc2 + bias + residual
//   one workgroup per 16 tokens; hidden (16x768) lives in LDS
// ---------------------------------------------------------------------------
__global__ __launch_bounds__(256)
void mlp_kernel(const _Float16* __restrict__ mn, const _Float16* __restrict__ w1,
                const float* __restrict__ fc1_b, const _Float16* __restrict__ w2,
                const float* __restrict__ fc2_b, const float* __restrict__ y1,
                float* __restrict__ out) {
  __shared__ __attribute__((aligned(16))) _Float16 Am[16 * 192];
  __shared__ __attribute__((aligned(16))) _Float16 Hs[16 * 768];
  __shared__ __attribute__((aligned(16))) _Float16 Bs[192 * 32];  // transposed
  const int tid = threadIdx.x, lane = tid & 31, wave = tid >> 5;
  const size_t r0 = (size_t)blockIdx.x * 16;

  for (int ch = tid; ch < 16 * 24; ch += 256)
    copy16_g2l(&mn[r0 * CC + ch * 8], &Am[ch * 8]);
  copy_join();
  __syncthreads();

  // fc1 + GELU
  for (int ct = 0; ct < 4; ++ct) {   // 4 column tiles of 192 over HIDDEN=768
    v8f acc[2] = {zero8(), zero8()};
    for (int kk = 0; kk < 192; kk += 32) {
      __syncthreads();
      for (int idx = tid; idx < 32 * 192; idx += 256) {
        const int kr = idx / 192, c = idx % 192;
        Bs[c * 32 + kr] = w1[(size_t)(kk + kr) * HIDDEN_ + ct * 192 + c];
      }
      __syncthreads();
#pragma unroll
      for (int j = 0; j < 2; ++j) {
        const int t = wave + 8 * j;          // 12 tiles over 16x192 (uniform)
        if (t < 12) {
          v16h a = load_frag(Am, 192, 0, kk, lane);
          v16h bb = load_frag(Bs, 32, t * 16, 0, lane);
          acc[j] = wmma16(a, bb, acc[j]);
        }
      }
    }
#pragma unroll
    for (int j = 0; j < 2; ++j) {
      const int t = wave + 8 * j;
      if (t < 12) {
#pragma unroll
        for (int e = 0; e < 8; ++e) {
          const int m = e + 8 * (lane >> 4);
          const int n = ct * 192 + t * 16 + (lane & 15);
          const float v = acc[j][e] + fc1_b[n];
          const float g = 0.5f * v * (1.f + erff(v * 0.70710678118654752f));
          Hs[m * HIDDEN_ + n] = (_Float16)g;
        }
      }
    }
  }
  __syncthreads();

  // fc2
  v8f acc2[2] = {zero8(), zero8()};
  for (int kk = 0; kk < HIDDEN_; kk += 32) {
    __syncthreads();
    for (int idx = tid; idx < 32 * 192; idx += 256) {
      const int kr = idx / 192, c = idx % 192;
      Bs[c * 32 + kr] = w2[(size_t)(kk + kr) * CC + c];
    }
    __syncthreads();
#pragma unroll
    for (int j = 0; j < 2; ++j) {
      const int t = wave + 8 * j;
      if (t < 12) {
        v16h a = load_frag(Hs, HIDDEN_, 0, kk, lane);
        v16h bb = load_frag(Bs, 32, t * 16, 0, lane);
        acc2[j] = wmma16(a, bb, acc2[j]);
      }
    }
  }
#pragma unroll
  for (int j = 0; j < 2; ++j) {
    const int t = wave + 8 * j;
    if (t < 12) {
#pragma unroll
      for (int e = 0; e < 8; ++e) {
        const int m = e + 8 * (lane >> 4);
        const int n = t * 16 + (lane & 15);
        const size_t o = (r0 + m) * CC + n;
        out[o] = y1[o] + acc2[j][e] + fc2_b[n];
      }
    }
  }
}

// ---------------------------------------------------------------------------
// Host launcher
// ---------------------------------------------------------------------------
extern "C" void kernel_launch(void* const* d_in, const int* in_sizes, int n_in,
                              void* d_out, int out_size, void* d_ws,
                              size_t ws_size, hipStream_t stream) {
  (void)in_sizes; (void)n_in; (void)out_size; (void)ws_size;

  const float* x       = (const float*)d_in[0];
  const float* norm1_w = (const float*)d_in[1];
  const float* norm1_b = (const float*)d_in[2];
  const float* qkv_w   = (const float*)d_in[3];
  const float* qkv_b   = (const float*)d_in[4];
  const float* rpb     = (const float*)d_in[5];
  const float* proj_w  = (const float*)d_in[6];
  const float* proj_b  = (const float*)d_in[7];
  const float* norm2_w = (const float*)d_in[8];
  const float* norm2_b = (const float*)d_in[9];
  const float* fc1_w   = (const float*)d_in[10];
  const float* fc1_b   = (const float*)d_in[11];
  const float* fc2_w   = (const float*)d_in[12];
  const float* fc2_b   = (const float*)d_in[13];
  float* out = (float*)d_out;

  char* ws = (char*)d_ws;
  size_t off = 0;
  auto alloc = [&](size_t bytes) {
    char* p = ws + off;
    off += (bytes + 255) & ~(size_t)255;
    return p;
  };
  _Float16* xn  = (_Float16*)alloc((size_t)NTOK * CC * 2);        // LN1 out
  float*    y1  = (float*)   alloc((size_t)NTOK * CC * 4);        // attn residual
  _Float16* mn  = (_Float16*)alloc((size_t)NTOK * CC * 2);        // LN2 out
  _Float16* ow  = (_Float16*)alloc((size_t)NWIN * NH_ * 64 * HD_ * 2); // attn O
  _Float16* wqh = (_Float16*)alloc((size_t)CC * 3 * CC * 2);      // qkv_w f16
  _Float16* wph = (_Float16*)alloc((size_t)CC * CC * 2);          // proj_w f16
  _Float16* w1h = (_Float16*)alloc((size_t)CC * HIDDEN_ * 2);     // fc1_w f16
  _Float16* w2h = (_Float16*)alloc((size_t)HIDDEN_ * CC * 2);     // fc2_w f16

  // 0) weight conversion
  const int nq = CC * 3 * CC, np = CC * CC, n1 = CC * HIDDEN_, n2 = HIDDEN_ * CC;
  cvt_f16_kernel<<<(nq + 255) / 256, 256, 0, stream>>>(qkv_w, wqh, nq);
  cvt_f16_kernel<<<(np + 255) / 256, 256, 0, stream>>>(proj_w, wph, np);
  cvt_f16_kernel<<<(n1 + 255) / 256, 256, 0, stream>>>(fc1_w, w1h, n1);
  cvt_f16_kernel<<<(n2 + 255) / 256, 256, 0, stream>>>(fc2_w, w2h, n2);

  // 1) LN1
  ln_f16_kernel<<<NTOK / 8, 256, 0, stream>>>(x, norm1_w, norm1_b, xn);
  // 2) fused window attention
  attn_kernel<<<NWIN, 256, 0, stream>>>(xn, wqh, qkv_b, rpb, ow);
  // 3) projection + residual (writes y1 = x + attn_out)
  proj_kernel<<<NWIN, 256, 0, stream>>>(ow, wph, proj_b, x, y1);
  // 4) LN2
  ln_f16_kernel<<<NTOK / 8, 256, 0, stream>>>(y1, norm2_w, norm2_b, mn);
  // 5) MLP + residual -> d_out
  mlp_kernel<<<NTOK / 16, 256, 0, stream>>>(mn, w1h, fc1_b, w2h, fc2_b, y1, out);
}
